// RNNvanilla_30537217474608
// MI455X (gfx1250) — compile-verified, matrix operand order checked by reading, not run on hardware
//
#include <hip/hip_runtime.h>
#include <hip/hip_bf16.h>

// ---------------------------------------------------------------------------
// LSTM classifier for MI455X (gfx1250): bf16 WMMA recurrence, persistent grid,
// parallel input-projection GEMM, async memory->LDS fills, cluster barrier.
// ---------------------------------------------------------------------------

typedef __bf16 bf16;
typedef __bf16 v16bf __attribute__((ext_vector_type(16)));
typedef __bf16 v8bf  __attribute__((ext_vector_type(8)));
typedef float  v8f   __attribute__((ext_vector_type(8)));
typedef int    v4i   __attribute__((ext_vector_type(4)));

#define NINP   256
#define NHID   512
#define BATCH  64
#define TLEN   512
#define KDIM   768            // NINP + NHID
#define NGATE  2048           // 4*NHID
#define KT     24             // KDIM/32
#define KT_X   8              // NINP/32  (input-projection K tiles)
#define NWG    16             // workgroups in persistent recurrent kernel
#define XSTR   776            // padded LDS row stride (bf16 elems), 16B aligned
#define ESTR   264            // padded emb row stride for k_gx (bf16)
#define GSTR   132            // padded gate LDS row stride (f32)
#define NFC1   170            // NHID // 3

#if __has_builtin(__builtin_amdgcn_global_load_async_to_lds_b128) && \
    __has_builtin(__builtin_amdgcn_s_wait_asynccnt)
#define HAVE_ASYNC_LDS 1
#else
#define HAVE_ASYNC_LDS 0
#endif

// float -> bf16 (RNE) via bit ops
__device__ __forceinline__ bf16 f2bf(float f) {
  unsigned u = __float_as_uint(f);
  unsigned r = (u + 0x7FFFu + ((u >> 16) & 1u)) >> 16;
  unsigned short us = (unsigned short)r;
  bf16 b;
  __builtin_memcpy(&b, &us, 2);
  return b;
}

// 16B memory -> LDS copy; async (ASYNCcnt) when available.
__device__ __forceinline__ void cp16(bf16* dst_lds, const bf16* src_gbl) {
#if HAVE_ASYNC_LDS
  __builtin_amdgcn_global_load_async_to_lds_b128(
      (v4i*)src_gbl, (v4i*)dst_lds, 0, 0);
#else
  *(v8bf*)dst_lds = *(const v8bf*)src_gbl;
#endif
}

__device__ __forceinline__ void cp_fence() {
#if HAVE_ASYNC_LDS
  __builtin_amdgcn_s_wait_asynccnt(0);
#endif
}

__device__ __forceinline__ void cluster_bar() {
#if __has_builtin(__builtin_amdgcn_s_cluster_barrier)
  __builtin_amdgcn_s_cluster_barrier();   // s_barrier_signal/-wait -3 (NOP if unclustered)
#endif
}

// K offset within a 32-wide K tile for the 16-bit WMMA A/B lane layout
__device__ __forceinline__ int k_off(int lane, int e) {
  return ((lane >> 4) & 1) * 8 + (e < 8 ? e : e + 8);
}

// --------------------------------------------------------------------------
// K0a: pack [W_ih | W_hh] (fp32 [2048 x 768]) into bf16 WMMA B-tiles:
// Wp[((ntile*KT + kt)*32 + lane)*16 + e]
// --------------------------------------------------------------------------
__global__ void k_pack(const float* __restrict__ Wih,
                       const float* __restrict__ Whh,
                       bf16* __restrict__ Wp) {
  int idx = blockIdx.x * 256 + threadIdx.x;
  if (idx >= NGATE * KDIM) return;
  int e    = idx & 15;
  int l    = (idx >> 4) & 31;
  int rest = idx >> 9;
  int kt   = rest % KT;
  int ntg  = rest / KT;
  int n = ntg * 16 + (l & 15);
  int k = kt * 32 + k_off(l, e);
  float v = (k < NINP) ? Wih[n * NINP + k] : Whh[n * NHID + (k - NINP)];
  Wp[idx] = f2bf(v);
}

// --------------------------------------------------------------------------
// K0b: misc init: h_buf[0] = bf16(hx); bias = b_ih + b_hh; counter = 0
// --------------------------------------------------------------------------
__global__ void k_misc(const float* __restrict__ hx,
                       const float* __restrict__ bih,
                       const float* __restrict__ bhh,
                       bf16* __restrict__ hbuf,
                       float* __restrict__ bias,
                       unsigned* __restrict__ counter) {
  int idx = blockIdx.x * 256 + threadIdx.x;
  if (idx < BATCH * NHID) {
    hbuf[idx] = f2bf(hx[idx]);
  } else if (idx < BATCH * NHID + NGATE) {
    int n = idx - BATCH * NHID;
    bias[n] = bih[n] + bhh[n];
  } else if (idx == BATCH * NHID + NGATE) {
    *counter = 0u;
  }
}

// --------------------------------------------------------------------------
// K1: embedding gather -> bf16  emb[t][b][k]
// --------------------------------------------------------------------------
__global__ void k_embed(const int* __restrict__ tok,
                        const float* __restrict__ enc,
                        bf16* __restrict__ emb) {
  int idx = blockIdx.x * 256 + threadIdx.x;   // over T*B*NINP/8
  int k8 = idx & 31;
  int b  = (idx >> 5) & 63;
  int t  = idx >> 11;
  int token = tok[b * TLEN + t];
  const float* src = enc + (size_t)token * NINP + k8 * 8;
  v8bf o;
#pragma unroll
  for (int i = 0; i < 8; ++i) o[i] = f2bf(src[i]);
  *(v8bf*)(emb + ((size_t)t * BATCH + b) * NINP + (size_t)k8 * 8) = o;
}

// --------------------------------------------------------------------------
// K_GX: parallel input projection Gx[t] = emb_t @ W_ih^T  (no seq dependency)
// M = T*B = 32768, N = 2048, K = 256; one WG per t; full-chip WMMA GEMM.
// Output pre-swizzled in C-tile lane layout:
//   gx[(((t*4 + mt)*128) + ntile)*256 + lane*8 + r]
// --------------------------------------------------------------------------
__launch_bounds__(256, 1)
__global__ void k_gx(const bf16* __restrict__ emb,
                     const bf16* __restrict__ Wp,
                     float* __restrict__ gx) {
  __shared__ __align__(16) bf16 lx[BATCH * ESTR];
  const int t = blockIdx.x;
  const int tid = threadIdx.x, lane = tid & 31, wv = tid >> 5;

  const bf16* embt = emb + (size_t)t * BATCH * NINP;
  for (int i = tid; i < BATCH * NINP / 8; i += 256) {
    int b = i >> 5, k8 = i & 31;
    cp16(&lx[b * ESTR + k8 * 8], &embt[b * NINP + k8 * 8]);
  }
  cp_fence();
  __syncthreads();

  const int aM = lane & 15, aKo = (lane >> 4) * 8;
  for (int nt = wv; nt < 128; nt += 8) {       // 16 N-tiles per wave
    const bf16* wp = Wp + (size_t)nt * KT * 512 + (size_t)lane * 16;
    v8f acc[4] = {};
#pragma unroll
    for (int kt = 0; kt < KT_X; ++kt) {
      v16bf bt = *(const v16bf*)wp;
      wp += 512;
      const int kb = kt * 32 + aKo;
#pragma unroll
      for (int mt = 0; mt < 4; ++mt) {
        const bf16* ap = &lx[(mt * 16 + aM) * ESTR + kb];
        union { v16bf v; v8bf h[2]; } au;
        au.h[0] = *(const v8bf*)ap;
        au.h[1] = *(const v8bf*)(ap + 16);
        acc[mt] = __builtin_amdgcn_wmma_f32_16x16x32_bf16(
            false, au.v, false, bt, (short)0, acc[mt], false, false);
      }
    }
#pragma unroll
    for (int mt = 0; mt < 4; ++mt)
      *(v8f*)(gx + ((((size_t)t * 4 + mt) * 128) + nt) * 256 + lane * 8) =
          acc[mt];
  }
}

// --------------------------------------------------------------------------
// K2: persistent recurrent kernel. 16 WGs x 256 threads (8 waves).
// use_gx=1: acc init from precomputed Gx, K-loop covers only W_hh (16 tiles).
// use_gx=0: fused path, K-loop covers [W_ih | W_hh] (24 tiles).
// --------------------------------------------------------------------------
__launch_bounds__(256, 1)
__global__ void k_rnn(const bf16* __restrict__ emb,
                      const float* __restrict__ gx, int use_gx,
                      const bf16* __restrict__ Wp,
                      const float* __restrict__ bias,
                      const float* __restrict__ cx,
                      const int* __restrict__ seq_len,
                      bf16* __restrict__ hbuf,
                      float* __restrict__ feat,
                      float* __restrict__ hx_out,
                      float* __restrict__ cx_out,
                      unsigned* __restrict__ counter) {
  __shared__ __align__(16) bf16  lds_x[BATCH * XSTR];
  __shared__ __align__(16) float lds_g[BATCH * GSTR];
  __shared__ __align__(16) float lds_c[BATCH * 32];
  __shared__ __align__(16) float lds_f[BATCH * 32];

  const int w    = blockIdx.x;
  const int tid  = threadIdx.x;
  const int lane = tid & 31;
  const int wv   = tid >> 5;

  for (int i = tid; i < BATCH * 32; i += 256) {
    int b = i >> 5, jl = i & 31;
    lds_c[i] = cx[b * NHID + w * 32 + jl];
    lds_f[i] = 0.f;
  }

  const int g   = wv >> 1;
  const int s   = wv & 1;
  const int ntg = g * 32 + w * 2 + s;
  const int kt0 = use_gx ? KT_X : 0;
  const bf16* wp_base = Wp + ((size_t)ntg * KT + kt0) * 512 + (size_t)lane * 16;

  const int aM   = lane & 15;
  const int aKo  = (lane >> 4) * 8;
  const int ccol = wv * 16 + (lane & 15);
  const int crow = (lane >> 4) * 8;

  for (int t = 0; t < TLEN; ++t) {
    // ---- fill LDS with x_t (emb part skipped when Gx is precomputed) ----
    if (!use_gx) {
      const bf16* embt = emb + (size_t)t * BATCH * NINP;
      for (int i = tid; i < BATCH * NINP / 8; i += 256) {
        int b = i >> 5, k8 = i & 31;
        cp16(&lds_x[b * XSTR + k8 * 8], &embt[b * NINP + k8 * 8]);
      }
    }
    const bf16* hsrc = hbuf + (size_t)(t & 1) * BATCH * NHID;
    for (int i = tid; i < BATCH * NHID / 8; i += 256) {
      int b = i >> 6, k8 = i & 63;
      cp16(&lds_x[b * XSTR + NINP + k8 * 8], &hsrc[b * NHID + k8 * 8]);
    }
    cp_fence();
    __syncthreads();

    // ---- accumulator init: Gx tile loads (split) or zeros (fused) ----
    v8f acc[4];
    if (use_gx) {
      const float* gxt =
          gx + (((size_t)t * 4) * 128 + ntg) * 256 + (size_t)lane * 8;
#pragma unroll
      for (int mt = 0; mt < 4; ++mt)
        acc[mt] = *(const v8f*)(gxt + (size_t)mt * 128 * 256);
    } else {
#pragma unroll
      for (int mt = 0; mt < 4; ++mt) acc[mt] = v8f{};
    }

    // ---- GEMM: 4 M-tiles x (24 - kt0) K-tiles of wmma bf16 ----
    const bf16* wp = wp_base;
#pragma unroll 4
    for (int kt = kt0; kt < KT; ++kt) {
      v16bf bt = *(const v16bf*)wp;
      __builtin_prefetch(wp + 512, 0, 3);
      wp += 512;
      const int kb = kt * 32 + aKo;
#pragma unroll
      for (int mt = 0; mt < 4; ++mt) {
        const bf16* ap = &lds_x[(mt * 16 + aM) * XSTR + kb];
        union { v16bf v; v8bf h[2]; } au;
        au.h[0] = *(const v8bf*)ap;
        au.h[1] = *(const v8bf*)(ap + 16);
        acc[mt] = __builtin_amdgcn_wmma_f32_16x16x32_bf16(
            false, au.v, false, bt, (short)0, acc[mt], false, false);
      }
    }

    // ---- scatter gate tiles to LDS ----
#pragma unroll
    for (int mt = 0; mt < 4; ++mt)
#pragma unroll
      for (int r = 0; r < 8; ++r)
        lds_g[(mt * 16 + crow + r) * GSTR + ccol] = acc[mt][r];
    __syncthreads();

    // ---- elementwise LSTM cell + h export + ragged pool ----
    bf16* hdst = hbuf + (size_t)((t + 1) & 1) * BATCH * NHID;
    for (int i = tid; i < BATCH * 32; i += 256) {
      int b = i >> 5, jl = i & 31;
      int j = w * 32 + jl;
      float gi = lds_g[b * GSTR + jl]      + bias[0 * NHID + j];
      float gf = lds_g[b * GSTR + 32 + jl] + bias[1 * NHID + j];
      float gg = lds_g[b * GSTR + 64 + jl] + bias[2 * NHID + j];
      float go = lds_g[b * GSTR + 96 + jl] + bias[3 * NHID + j];
      gi = 1.f / (1.f + __expf(-gi));
      gf = 1.f / (1.f + __expf(-gf));
      go = 1.f / (1.f + __expf(-go));
      gg = tanhf(gg);
      float c_new = gf * lds_c[i] + gi * gg;
      float h_new = go * tanhf(c_new);
      lds_c[i] = c_new;
      hdst[b * NHID + j] = f2bf(h_new);
      if (t < seq_len[b]) lds_f[i] += h_new;
      if (t == TLEN - 1) {
        hx_out[b * NHID + j] = h_new;
        cx_out[b * NHID + j] = c_new;
      }
    }

    // ---- grid barrier: cluster barrier (hw fast path) + atomic counter ----
    __threadfence();
    __syncthreads();
    cluster_bar();
    if (tid == 0) {
      __hip_atomic_fetch_add(counter, 1u, __ATOMIC_RELEASE,
                             __HIP_MEMORY_SCOPE_AGENT);
      const unsigned want = (unsigned)(NWG * (t + 1));
      while (__hip_atomic_load(counter, __ATOMIC_ACQUIRE,
                               __HIP_MEMORY_SCOPE_AGENT) < want)
        __builtin_amdgcn_s_sleep(2);
    }
    __syncthreads();
  }

  for (int i = tid; i < BATCH * 32; i += 256) {
    int b = i >> 5, jl = i & 31;
    feat[b * NHID + w * 32 + jl] = lds_f[i];
  }
}

// --------------------------------------------------------------------------
// K3: head: feature = feat/seq_len; fc1; fc2; log_softmax.
// --------------------------------------------------------------------------
__global__ void k_head(const float* __restrict__ feat,
                       const int* __restrict__ seq_len,
                       const float* __restrict__ fc1w,
                       const float* __restrict__ fc1b,
                       const float* __restrict__ fc2w,
                       const float* __restrict__ fc2b,
                       float* __restrict__ out_logits) {
  __shared__ float h1[BATCH * (NFC1 + 1)];
  int tid = threadIdx.x;
  for (int i = tid; i < BATCH * NFC1; i += 256) {
    int b = i / NFC1, r = i % NFC1;
    const float* fv = feat + b * NHID;
    const float* wv = fc1w + r * NHID;
    float acc = 0.f;
#pragma unroll 4
    for (int k = 0; k < NHID; ++k) acc += fv[k] * wv[k];
    h1[b * (NFC1 + 1) + r] = acc / (float)seq_len[b] + fc1b[r];
  }
  __syncthreads();
  if (tid < BATCH) {
    int b = tid;
    float l0 = fc2b[0], l1 = fc2b[1];
    for (int r = 0; r < NFC1; ++r) {
      float h = h1[b * (NFC1 + 1) + r];
      l0 += h * fc2w[r];
      l1 += h * fc2w[NFC1 + r];
    }
    float m = fmaxf(l0, l1);
    float lse = m + logf(__expf(l0 - m) + __expf(l1 - m));
    out_logits[b * 2 + 0] = l0 - lse;
    out_logits[b * 2 + 1] = l1 - lse;
  }
}

// --------------------------------------------------------------------------
extern "C" void kernel_launch(void* const* d_in, const int* in_sizes, int n_in,
                              void* d_out, int out_size, void* d_ws,
                              size_t ws_size, hipStream_t stream) {
  const int*   input  = (const int*)  d_in[0];
  const float* hx     = (const float*)d_in[1];
  const float* cx     = (const float*)d_in[2];
  const int*   seqlen = (const int*)  d_in[3];
  const float* enc    = (const float*)d_in[4];
  const float* Wih    = (const float*)d_in[5];
  const float* Whh    = (const float*)d_in[6];
  const float* bih    = (const float*)d_in[7];
  const float* bhh    = (const float*)d_in[8];
  const float* fc1w   = (const float*)d_in[9];
  const float* fc1b   = (const float*)d_in[10];
  const float* fc2w   = (const float*)d_in[11];
  const float* fc2b   = (const float*)d_in[12];

  char* ws0 = (char*)d_ws;
  char* ws  = ws0;
  bf16*     Wp      = (bf16*)ws;      ws += (size_t)NGATE * KDIM * 2;        // 3 MB
  float*    bias    = (float*)ws;     ws += (size_t)NGATE * 4;               // 8 KB
  bf16*     embb    = (bf16*)ws;      ws += (size_t)TLEN * BATCH * NINP * 2; // 16 MB
  bf16*     hbuf    = (bf16*)ws;      ws += (size_t)2 * BATCH * NHID * 2;    // 128 KB
  float*    feat    = (float*)ws;     ws += (size_t)BATCH * NHID * 4;        // 128 KB
  unsigned* counter = (unsigned*)ws;  ws += 256;
  float*    gxbuf   = (float*)ws;     // 268 MB when available
  const size_t gx_bytes = (size_t)TLEN * BATCH * NGATE * 4;
  const size_t base_need = (size_t)(ws - ws0);
  const int use_gx = (ws_size >= base_need + gx_bytes) ? 1 : 0;

  float* out    = (float*)d_out;
  float* logits = out;                       // [64,2]
  float* hx_out = out + BATCH * 2;           // [64,512]
  float* cx_out = hx_out + BATCH * NHID;     // [64,512]

  k_pack<<<(NGATE * KDIM + 255) / 256, 256, 0, stream>>>(Wih, Whh, Wp);
  k_misc<<<(BATCH * NHID + NGATE + 1 + 255) / 256, 256, 0, stream>>>(
      hx, bih, bhh, hbuf, bias, counter);
  k_embed<<<(TLEN * BATCH * NINP / 8) / 256, 256, 0, stream>>>(input, enc, embb);
  if (use_gx)
    k_gx<<<TLEN, 256, 0, stream>>>(embb, Wp, gxbuf);
  k_rnn<<<NWG, 256, 0, stream>>>(embb, gxbuf, use_gx, Wp, bias, cx, seqlen,
                                 hbuf, feat, hx_out, cx_out, counter);
  k_head<<<1, 256, 0, stream>>>(feat, seqlen, fc1w, fc1b, fc2w, fc2b, logits);
}
